// RAGMaster_17403207483535
// MI455X (gfx1250) — matrix operand order repaired and checked by reading, not compiled, not run on hardware
//
#include <hip/hip_runtime.h>

typedef __attribute__((ext_vector_type(16))) __bf16 v16bf;
typedef __attribute__((ext_vector_type(8)))  float  v8f;
typedef __attribute__((ext_vector_type(2)))  float  v2f;

#define NB      1024     // batch rows
#define NH      100000   // hist rows
#define NPADH   100096   // padded to 256*391
#define KD      158
#define KP      160      // K padded to 5*32
#define NCHUNKS 16
#define CHUNK   6256     // 391 tiles * 16
#define TILES   391
#define TVALID  385      // tiles 385..390 of the last chunk are pure padding
#define DM      256
#define XROW    1768     // 8*221
#define XLAST   1547     // 7*221
#define NEG_INF (-3.0e38f)

// ---------------- device helpers ----------------
__device__ __forceinline__ float wredsum(float v) {
#pragma unroll
  for (int o = 16; o; o >>= 1) v += __shfl_xor(v, o);
  return v;
}
__device__ __forceinline__ float wredmax(float v) {
#pragma unroll
  for (int o = 16; o; o >>= 1) v = fmaxf(v, __shfl_xor(v, o));
  return v;
}
// LayerNorm one row, one wave (wave-synchronous, no block barrier)
__device__ __forceinline__ void ln_row(float* rowp, int W, const float* g, const float* b, int lane) {
  float s = 0.f;
  for (int c = lane; c < W; c += 32) s += rowp[c];
  s = wredsum(s);
  float mean = s / (float)W;
  float v = 0.f;
  for (int c = lane; c < W; c += 32) { float d = rowp[c] - mean; v += d * d; }
  v = wredsum(v) / (float)W;
  float inv = rsqrtf(v + 1e-5f);
  for (int c = lane; c < W; c += 32) rowp[c] = (rowp[c] - mean) * inv * g[c] + b[c];
}
// register-resident sorted-descending top-5 insert
__device__ __forceinline__ void ins5(float v, int id, float tv[5], int ti[5]) {
  if (v <= tv[4]) return;
  tv[4] = v; ti[4] = id;
#pragma unroll
  for (int k = 4; k > 0; --k) {
    if (tv[k] > tv[k - 1]) {
      float fv = tv[k]; tv[k] = tv[k - 1]; tv[k - 1] = fv;
      int   fi = ti[k]; ti[k] = ti[k - 1]; ti[k - 1] = fi;
    }
  }
}

// ---------------- kernel 1: normalize hist -> bf16, pad K and N ----------------
__global__ __launch_bounds__(256) void hist_prep(const float* __restrict__ hist, __bf16* __restrict__ hn) {
  int row  = blockIdx.x * 8 + (threadIdx.x >> 5);
  int lane = threadIdx.x & 31;
  if (row >= NPADH) return;
  if (row < NH) {
    float s = 0.f;
    for (int c = lane; c < KD; c += 32) { float v = hist[(size_t)row * KD + c]; s += v * v; }
    s = wredsum(s);
    float inv = 1.f / fmaxf(sqrtf(s), 1e-8f);
    for (int c = lane; c < KP; c += 32) {
      float v = (c < KD) ? hist[(size_t)row * KD + c] * inv : 0.f;
      hn[(size_t)row * KP + c] = (__bf16)v;
    }
  } else {
    for (int c = lane; c < KP; c += 32) hn[(size_t)row * KP + c] = (__bf16)0.f;
  }
}

// ---------------- kernel 2: extract xl, norms, bf16 copy ----------------
__global__ __launch_bounds__(256) void xl_prep(const float* __restrict__ x, float* __restrict__ xlf,
                                               __bf16* __restrict__ xlbf, float* __restrict__ xln) {
  int row  = blockIdx.x * 8 + (threadIdx.x >> 5);
  int lane = threadIdx.x & 31;
  if (row >= NB) return;
  const float* xr = x + (size_t)row * XROW + XLAST;
  float s = 0.f;
  for (int c = lane; c < KD; c += 32) { float v = xr[c]; s += v * v; }
  s = wredsum(s);
  float nrm = fmaxf(sqrtf(s), 1e-8f);
  if (lane == 0) xln[row] = nrm;
  for (int c = lane; c < KP; c += 32) {
    float v = (c < KD) ? xr[c] : 0.f;
    if (c < KD) xlf[(size_t)row * KD + c] = v;
    xlbf[(size_t)row * KP + c] = (__bf16)v;
  }
}

// ---------------- kernel 3: fused bf16-WMMA GEMM + per-row top-5 ----------------
__global__ __launch_bounds__(256) void sims_topk(const __bf16* __restrict__ hn, const __bf16* __restrict__ xlbf,
                                                 float* __restrict__ cv, int* __restrict__ ci) {
  __shared__ __bf16 s_xl[16 * KP];
  __shared__ float  s_c[8][16][16];
  const int tid = threadIdx.x, lane = tid & 31, wv = tid >> 5;
  const int bm0 = blockIdx.x * 16, chunk = blockIdx.y;

  for (int i = tid; i < 16 * KP; i += 256) s_xl[i] = xlbf[(size_t)bm0 * KP + i];
  __syncthreads();

  // A fragments (16x32 bf16 per K-step), contiguous k = 16*(lane/16) + s, matching B
  v16bf A[5];
  const int arow = lane & 15;
  const int aoff = (lane < 16) ? 0 : 16;
#pragma unroll
  for (int ks = 0; ks < 5; ++ks) {
#pragma unroll
    for (int e = 0; e < 8; ++e) {
      A[ks][e]     = s_xl[arow * KP + ks * 32 + aoff + e];
      A[ks][8 + e] = s_xl[arow * KP + ks * 32 + aoff + 8 + e];
    }
  }

  float tv[5]; int ti[5];
#pragma unroll
  for (int k = 0; k < 5; ++k) { tv[k] = NEG_INF; ti[k] = 0; }

  const int srow = lane & 15;
  const int col0 = (lane >> 4) * 8;
  const int jcol = lane & 15;
  const int koff = (lane & 16) ? 16 : 0;
  // last 6 tiles of the last chunk are pure padding -> clamp, no per-column check
  const int tend = (chunk == NCHUNKS - 1) ? TVALID : TILES;

  for (int t = wv; t < tend; t += 8) {
    const int nbase = chunk * CHUNK + t * 16;
    const __bf16* bp = hn + (size_t)(nbase + jcol) * KP + koff;
    v8f acc = {};
#pragma unroll
    for (int ks = 0; ks < 5; ++ks) {
      v16bf Bf = *reinterpret_cast<const v16bf*>(bp + ks * 32);
      acc = __builtin_amdgcn_wmma_f32_16x16x32_bf16(false, A[ks], false, Bf, (short)0, acc, false, false);
    }
    // scatter C tile to this wave's LDS region (in-order DS within wave)
#pragma unroll
    for (int r = 0; r < 8; ++r) {
      int m = r + ((lane >> 4) << 3);
      s_c[wv][m][lane & 15] = acc[r];
    }
    __builtin_amdgcn_wave_barrier();
    {
      const float4* p = reinterpret_cast<const float4*>(&s_c[wv][srow][col0]);
      float4 u0 = p[0], u1 = p[1];
      float vv[8] = {u0.x, u0.y, u0.z, u0.w, u1.x, u1.y, u1.z, u1.w};
#pragma unroll
      for (int c = 0; c < 8; ++c) ins5(vv[c], nbase + col0 + c, tv, ti);
    }
    __builtin_amdgcn_wave_barrier();
  }

  const int rowg = bm0 + srow, half = lane >> 4;
  size_t base = (((size_t)rowg * NCHUNKS + chunk) * 8 + wv) * 10 + half * 5;
#pragma unroll
  for (int k = 0; k < 5; ++k) { cv[base + k] = tv[k]; ci[base + k] = ti[k]; }
}

// ---------------- kernel 4: merge candidates -> top-5 + softmax weights ----------------
__global__ void topk_merge(const float* __restrict__ cv, const int* __restrict__ ci,
                           const float* __restrict__ xln, float* __restrict__ tw, int* __restrict__ tix) {
  const int row = blockIdx.x, lane = threadIdx.x;
  float lv[5]; int li[5];
#pragma unroll
  for (int k = 0; k < 5; ++k) { lv[k] = NEG_INF; li[k] = 0; }
  size_t base = (size_t)row * (NCHUNKS * 8 * 10);
  for (int k = lane; k < NCHUNKS * 8 * 10; k += 32) ins5(cv[base + k], ci[base + k], lv, li);

  float outv[5]; int outi[5];
#pragma unroll
  for (int r = 0; r < 5; ++r) {
    float v = lv[0]; int id = li[0]; int src = lane;
#pragma unroll
    for (int o = 16; o; o >>= 1) {
      float ov = __shfl_xor(v, o); int oid = __shfl_xor(id, o); int os = __shfl_xor(src, o);
      if (ov > v || (ov == v && os < src)) { v = ov; id = oid; src = os; }
    }
    if (lane == src) {
#pragma unroll
      for (int k = 0; k < 4; ++k) { lv[k] = lv[k + 1]; li[k] = li[k + 1]; }
      lv[4] = NEG_INF; li[4] = 0;
    }
    outv[r] = v; outi[r] = id;
  }
  if (lane == 0) {
    float inv_n = 1.f / xln[row];
    float m = outv[0] * inv_n, s = 0.f, e[5];
#pragma unroll
    for (int k = 0; k < 5; ++k) { e[k] = expf(outv[k] * inv_n - m); s += e[k]; }
#pragma unroll
    for (int k = 0; k < 5; ++k) { tw[row * 5 + k] = e[k] / s; tix[row * 5 + k] = outi[k]; }
  }
}

// ---------------- kernel 5: aug = [xl, wsum x5] ----------------
__global__ __launch_bounds__(256) void build_aug(const float* __restrict__ hist, const float* __restrict__ xlf,
                                                 const float* __restrict__ tw, const int* __restrict__ tix,
                                                 float* __restrict__ aug) {
  const int row = blockIdx.x, tid = threadIdx.x;
  __shared__ float w[5]; __shared__ int id[5];
  if (tid < 5) { w[tid] = tw[row * 5 + tid]; id[tid] = tix[row * 5 + tid]; }
  __syncthreads();
  for (int c = tid; c < KD; c += 256) {
    float ws = 0.f;
#pragma unroll
    for (int k = 0; k < 5; ++k) ws += w[k] * hist[(size_t)id[k] * KD + c];
    aug[(size_t)row * 948 + c] = xlf[(size_t)row * KD + c];
#pragma unroll
    for (int j = 1; j < 6; ++j) aug[(size_t)row * 948 + j * KD + c] = ws;
  }
}

// ---------------- kernel 6: per-row MLP/gate/temporal chain -> zn2 ----------------
__global__ __launch_bounds__(256) void mlp_chain(
    const float* __restrict__ aug, const float* __restrict__ x,
    const float* __restrict__ fW1, const float* __restrict__ fb1,
    const float* __restrict__ fW2, const float* __restrict__ fb2,
    const float* __restrict__ flg, const float* __restrict__ flb,
    const float* __restrict__ gW, const float* __restrict__ gb,
    const float* __restrict__ inW, const float* __restrict__ inb,
    const float* __restrict__ tWv,
    const float* __restrict__ tl1g, const float* __restrict__ tl1b,
    const float* __restrict__ tl2g, const float* __restrict__ tl2b,
    const float* __restrict__ tfW1, const float* __restrict__ tfb1,
    const float* __restrict__ tfW2, const float* __restrict__ tfb2,
    const float* __restrict__ sl1g, const float* __restrict__ sl1b,
    float* __restrict__ zn2) {
  __shared__ float s_h1[8][316];
  __shared__ float s_f[8][158];
  __shared__ float s_g[8][158];
  __shared__ float s_a[8][256];
  __shared__ float s_b[8][256];
  __shared__ float s_c[8][256];
  const int tid = threadIdx.x, lane = tid & 31, wvid = tid >> 5;
  const int r0 = blockIdx.x * 8;

  // h1 = relu(aug @ fW1 + fb1)
  for (int j = tid; j < 316; j += 256) {
    float acc[8];
#pragma unroll
    for (int r = 0; r < 8; ++r) acc[r] = fb1[j];
    for (int i = 0; i < 948; ++i) {
      float wv = fW1[i * 316 + j];
#pragma unroll
      for (int r = 0; r < 8; ++r) acc[r] += aug[(size_t)(r0 + r) * 948 + i] * wv;
    }
#pragma unroll
    for (int r = 0; r < 8; ++r) s_h1[r][j] = fmaxf(acc[r], 0.f);
  }
  __syncthreads();
  // f = h1 @ fW2 + fb2
  for (int j = tid; j < 158; j += 256) {
    float acc[8];
#pragma unroll
    for (int r = 0; r < 8; ++r) acc[r] = fb2[j];
    for (int i = 0; i < 316; ++i) {
      float wv = fW2[i * 158 + j];
#pragma unroll
      for (int r = 0; r < 8; ++r) acc[r] += s_h1[r][i] * wv;
    }
#pragma unroll
    for (int r = 0; r < 8; ++r) s_f[r][j] = acc[r];
  }
  __syncthreads();
  ln_row(&s_f[wvid][0], 158, flg, flb, lane);
  __syncthreads();
  // gate logits
  for (int j = tid; j < 158; j += 256) {
    float acc[8];
#pragma unroll
    for (int r = 0; r < 8; ++r) acc[r] = gb[j];
    for (int i = 0; i < 63; ++i) {
      float wv = gW[i * 158 + j];
#pragma unroll
      for (int r = 0; r < 8; ++r) acc[r] += x[(size_t)(r0 + r) * XROW + XLAST + 158 + i] * wv;
    }
#pragma unroll
    for (int r = 0; r < 8; ++r) s_g[r][j] = acc[r];
  }
  __syncthreads();
  // gw = 158 * softmax(gate); zin = fused * gw
  {
    float* grow = &s_g[wvid][0]; float* frow = &s_f[wvid][0];
    float m = NEG_INF;
    for (int c = lane; c < 158; c += 32) m = fmaxf(m, grow[c]);
    m = wredmax(m);
    float s = 0.f;
    for (int c = lane; c < 158; c += 32) { float e = expf(grow[c] - m); grow[c] = e; s += e; }
    s = wredsum(s);
    float k = 158.f / s;
    for (int c = lane; c < 158; c += 32) frow[c] *= grow[c] * k;
  }
  __syncthreads();
  // z0 = zin @ inW + inb + pe(odd += 1)
  for (int j = tid; j < 256; j += 256) {
    float acc[8];
#pragma unroll
    for (int r = 0; r < 8; ++r) acc[r] = inb[j] + ((j & 1) ? 1.f : 0.f);
    for (int i = 0; i < 158; ++i) {
      float wv = inW[i * 256 + j];
#pragma unroll
      for (int r = 0; r < 8; ++r) acc[r] += s_f[r][i] * wv;
    }
#pragma unroll
    for (int r = 0; r < 8; ++r) s_a[r][j] = acc[r];
  }
  __syncthreads();
  ln_row(&s_a[wvid][0], 256, tl1g, tl1b, lane);   // zn
  __syncthreads();
  // xt = LN(zn + zn@tWv)
  for (int j = tid; j < 256; j += 256) {
    float acc[8];
#pragma unroll
    for (int r = 0; r < 8; ++r) acc[r] = 0.f;
    for (int i = 0; i < 256; ++i) {
      float wv = tWv[i * 256 + j];
#pragma unroll
      for (int r = 0; r < 8; ++r) acc[r] += s_a[r][i] * wv;
    }
#pragma unroll
    for (int r = 0; r < 8; ++r) s_b[r][j] = s_a[r][j] + acc[r];
  }
  __syncthreads();
  ln_row(&s_b[wvid][0], 256, tl2g, tl2b, lane);   // xt
  __syncthreads();
  // ffn: h = relu(xt@tfW1+tfb1)
  for (int j = tid; j < 256; j += 256) {
    float acc[8];
#pragma unroll
    for (int r = 0; r < 8; ++r) acc[r] = tfb1[j];
    for (int i = 0; i < 256; ++i) {
      float wv = tfW1[i * 256 + j];
#pragma unroll
      for (int r = 0; r < 8; ++r) acc[r] += s_b[r][i] * wv;
    }
#pragma unroll
    for (int r = 0; r < 8; ++r) s_c[r][j] = fmaxf(acc[r], 0.f);
  }
  __syncthreads();
  // z1 = xt + h@tfW2 + tfb2
  for (int j = tid; j < 256; j += 256) {
    float acc[8];
#pragma unroll
    for (int r = 0; r < 8; ++r) acc[r] = tfb2[j];
    for (int i = 0; i < 256; ++i) {
      float wv = tfW2[i * 256 + j];
#pragma unroll
      for (int r = 0; r < 8; ++r) acc[r] += s_c[r][i] * wv;
    }
#pragma unroll
    for (int r = 0; r < 8; ++r) s_a[r][j] = s_b[r][j] + acc[r];
  }
  __syncthreads();
  ln_row(&s_a[wvid][0], 256, sl1g, sl1b, lane);   // zn2
  __syncthreads();
  for (int j = tid; j < 256; j += 256) {
#pragma unroll
    for (int r = 0; r < 8; ++r) zn2[(size_t)(r0 + r) * 256 + j] = s_a[r][j];
  }
}

// ---------------- kernel 7: q,k,v = zn2 @ sW{q,k,v} via f32 WMMA ----------------
__global__ __launch_bounds__(256) void qkv_wmma(const float* __restrict__ zn2,
    const float* __restrict__ sWq, const float* __restrict__ sWk, const float* __restrict__ sWv,
    float* __restrict__ q, float* __restrict__ k, float* __restrict__ v) {
  __shared__ float s_z[16][260];   // pad: stride 260 -> row gathers land on distinct banks
  const int tid = threadIdx.x, lane = tid & 31, wv = tid >> 5;
  const int m0 = blockIdx.x * 16;
  for (int i = tid; i < 4096; i += 256) s_z[i >> 8][i & 255] = zn2[(size_t)m0 * 256 + i];
  __syncthreads();
  const int mn = lane & 15, ksel = (lane >> 4) * 2;
  for (int task = wv; task < 48; task += 8) {
    const int mat = task >> 4, n0 = (task & 15) * 16;
    const float* W = (mat == 0) ? sWq : ((mat == 1) ? sWk : sWv);
    float* O = (mat == 0) ? q : ((mat == 1) ? k : v);
    v8f acc = {};
    for (int ks = 0; ks < 64; ++ks) {
      const int kk = ks * 4 + ksel;
      v2f a, b;
      a[0] = s_z[mn][kk];
      a[1] = s_z[mn][kk + 1];
      b[0] = W[(size_t)kk * 256 + n0 + mn];
      b[1] = W[(size_t)(kk + 1) * 256 + n0 + mn];
      acc = __builtin_amdgcn_wmma_f32_16x16x4_f32(false, a, false, b, (short)0, acc, false, false);
    }
#pragma unroll
    for (int r = 0; r < 8; ++r) {
      int mm = r + ((lane >> 4) << 3);
      O[(size_t)(m0 + mm) * 256 + n0 + mn] = acc[r];
    }
  }
}

// ---------------- kernel 8: attention scores via f32 WMMA ----------------
__global__ __launch_bounds__(256) void scores_wmma(const float* __restrict__ q, const float* __restrict__ k,
                                                   float* __restrict__ sc) {
  __shared__ float s_q[16][132];   // pad 132 -> conflict-free row gathers
  const int q0 = blockIdx.x * 16, h = blockIdx.y;
  const int tid = threadIdx.x, lane = tid & 31, wv = tid >> 5;
  for (int i = tid; i < 2048; i += 256) {
    int r = i >> 7, d = i & 127;
    s_q[r][d] = q[(size_t)(q0 + r) * 256 + h * 128 + d];
  }
  __syncthreads();
  const int mn = lane & 15, ksel = (lane >> 4) * 2;
  for (int kt = wv; kt < 64; kt += 8) {
    const int k0 = kt * 16;
    const float* kbase = k + (size_t)(k0 + mn) * 256 + h * 128;
    v8f acc = {};
    for (int ks = 0; ks < 32; ++ks) {
      const int kk = ks * 4 + ksel;
      v2f a, b;
      a[0] = s_q[mn][kk];
      a[1] = s_q[mn][kk + 1];
      b[0] = kbase[kk];
      b[1] = kbase[kk + 1];
      acc = __builtin_amdgcn_wmma_f32_16x16x4_f32(false, a, false, b, (short)0, acc, false, false);
    }
#pragma unroll
    for (int r = 0; r < 8; ++r) {
      int mm = r + ((lane >> 4) << 3);
      sc[((size_t)(h * 1024 + q0 + mm)) * 1024 + k0 + mn] = acc[r] * 0.08838834764831845f;
    }
  }
}

// ---------------- kernel 9: row softmax over 1024 ----------------
__global__ __launch_bounds__(256) void softmax1024(float* __restrict__ sc) {
  __shared__ float red[256];
  const int row = blockIdx.x, tid = threadIdx.x;
  const size_t base = (size_t)row * 1024;
  float v[4];
#pragma unroll
  for (int kk = 0; kk < 4; ++kk) v[kk] = sc[base + tid + 256 * kk];
  float m = fmaxf(fmaxf(v[0], v[1]), fmaxf(v[2], v[3]));
  red[tid] = m; __syncthreads();
  for (int s = 128; s; s >>= 1) { if (tid < s) red[tid] = fmaxf(red[tid], red[tid + s]); __syncthreads(); }
  m = red[0]; __syncthreads();
  float sum = 0.f;
#pragma unroll
  for (int kk = 0; kk < 4; ++kk) { v[kk] = expf(v[kk] - m); sum += v[kk]; }
  red[tid] = sum; __syncthreads();
  for (int s = 128; s; s >>= 1) { if (tid < s) red[tid] += red[tid + s]; __syncthreads(); }
  float inv = 1.f / red[0];
#pragma unroll
  for (int kk = 0; kk < 4; ++kk) sc[base + tid + 256 * kk] = v[kk] * inv;
}

// ---------------- kernel 10: att = A @ V via f32 WMMA ----------------
__global__ __launch_bounds__(256) void av_wmma(const float* __restrict__ sc, const float* __restrict__ v,
                                               float* __restrict__ att) {
  const int q0 = blockIdx.x * 16, h = blockIdx.y;
  const int tid = threadIdx.x, lane = tid & 31, wv = tid >> 5;
  const int mn = lane & 15, ksel = (lane >> 4) * 2;
  const int d0 = wv * 16;                       // wave owns one 16-wide slice of dh=128
  const float* abase = sc + ((size_t)(h * 1024 + q0 + mn)) * 1024;
  v8f acc = {};
  for (int ks = 0; ks < 256; ++ks) {
    const int kk = ks * 4 + ksel;
    v2f a, b;
    a[0] = abase[kk];
    a[1] = abase[kk + 1];
    b[0] = v[(size_t)kk * 256 + h * 128 + d0 + mn];
    b[1] = v[(size_t)(kk + 1) * 256 + h * 128 + d0 + mn];
    acc = __builtin_amdgcn_wmma_f32_16x16x4_f32(false, a, false, b, (short)0, acc, false, false);
  }
#pragma unroll
  for (int r = 0; r < 8; ++r) {
    int mm = r + ((lane >> 4) << 3);
    att[(size_t)h * 131072 + (size_t)(q0 + mm) * 128 + d0 + mn] = acc[r];
  }
}

// ---------------- kernel 11: LN + spatial FFN + output head ----------------
__global__ __launch_bounds__(256) void final_kernel(const float* __restrict__ zn2, const float* __restrict__ att,
    const float* __restrict__ sl2g, const float* __restrict__ sl2b,
    const float* __restrict__ sfW1, const float* __restrict__ sfb1,
    const float* __restrict__ sfW2, const float* __restrict__ sfb2,
    const float* __restrict__ oW, const float* __restrict__ ob, float* __restrict__ out) {
  __shared__ float s_x[8][256];
  __shared__ float s_h[8][256];
  __shared__ float s_z[8][256];
  const int tid = threadIdx.x, lane = tid & 31, wvid = tid >> 5;
  const int r0 = blockIdx.x * 8;
  for (int j = tid; j < 256; j += 256) {
#pragma unroll
    for (int r = 0; r < 8; ++r)
      s_x[r][j] = zn2[(size_t)(r0 + r) * 256 + j] +
                  att[(size_t)(j >> 7) * 131072 + (size_t)(r0 + r) * 128 + (j & 127)];
  }
  __syncthreads();
  ln_row(&s_x[wvid][0], 256, sl2g, sl2b, lane);   // xt2
  __syncthreads();
  for (int j = tid; j < 256; j += 256) {
    float acc[8];
#pragma unroll
    for (int r = 0; r < 8; ++r) acc[r] = sfb1[j];
    for (int i = 0; i < 256; ++i) { float wv = sfW1[i * 256 + j];
#pragma unroll
      for (int r = 0; r < 8; ++r) acc[r] += s_x[r][i] * wv; }
#pragma unroll
    for (int r = 0; r < 8; ++r) s_h[r][j] = fmaxf(acc[r], 0.f);
  }
  __syncthreads();
  for (int j = tid; j < 256; j += 256) {
    float acc[8];
#pragma unroll
    for (int r = 0; r < 8; ++r) acc[r] = sfb2[j];
    for (int i = 0; i < 256; ++i) { float wv = sfW2[i * 256 + j];
#pragma unroll
      for (int r = 0; r < 8; ++r) acc[r] += s_h[r][i] * wv; }
#pragma unroll
    for (int r = 0; r < 8; ++r) s_z[r][j] = s_x[r][j] + acc[r];
  }
  __syncthreads();
  float o = 0.f;
  for (int c = lane; c < 256; c += 32) o += s_z[wvid][c] * oW[c];
  o = wredsum(o);
  if (lane == 0) out[r0 + wvid] = o + ob[0];
}

// ---------------- workspace layout ----------------
static constexpr size_t al256(size_t x) { return (x + 255) & ~(size_t)255; }
static constexpr size_t OFF_HN   = 0;
static constexpr size_t OFF_XLF  = al256(OFF_HN   + (size_t)NPADH * KP * 2);
static constexpr size_t OFF_XLBF = al256(OFF_XLF  + (size_t)NB * KD * 4);
static constexpr size_t OFF_XLN  = al256(OFF_XLBF + (size_t)NB * KP * 2);
static constexpr size_t OFF_CV   = al256(OFF_XLN  + (size_t)NB * 4);
static constexpr size_t OFF_CI   = al256(OFF_CV   + (size_t)NB * NCHUNKS * 8 * 10 * 4);
static constexpr size_t OFF_TW   = al256(OFF_CI   + (size_t)NB * NCHUNKS * 8 * 10 * 4);
static constexpr size_t OFF_TI   = al256(OFF_TW   + (size_t)NB * 5 * 4);
static constexpr size_t OFF_AUG  = al256(OFF_TI   + (size_t)NB * 5 * 4);
static constexpr size_t OFF_ZN2  = al256(OFF_AUG  + (size_t)NB * 948 * 4);
static constexpr size_t OFF_Q    = al256(OFF_ZN2  + (size_t)NB * DM * 4);
static constexpr size_t OFF_K    = al256(OFF_Q    + (size_t)NB * DM * 4);
static constexpr size_t OFF_V    = al256(OFF_K    + (size_t)NB * DM * 4);
static constexpr size_t OFF_SC   = al256(OFF_V    + (size_t)NB * DM * 4);
static constexpr size_t OFF_ATT  = al256(OFF_SC   + (size_t)2 * 1024 * 1024 * 4);

extern "C" void kernel_launch(void* const* d_in, const int* in_sizes, int n_in,
                              void* d_out, int out_size, void* d_ws, size_t ws_size,
                              hipStream_t stream) {
  (void)in_sizes; (void)n_in; (void)out_size; (void)ws_size;
  const float* x    = (const float*)d_in[0];
  const float* hist = (const float*)d_in[1];
  const float* fW1  = (const float*)d_in[2];
  const float* fb1  = (const float*)d_in[3];
  const float* fW2  = (const float*)d_in[4];
  const float* fb2  = (const float*)d_in[5];
  const float* flg  = (const float*)d_in[6];
  const float* flb  = (const float*)d_in[7];
  const float* gW   = (const float*)d_in[8];
  const float* gb   = (const float*)d_in[9];
  const float* inW  = (const float*)d_in[10];
  const float* inb  = (const float*)d_in[11];
  const float* tWv  = (const float*)d_in[14];
  const float* tl1g = (const float*)d_in[15];
  const float* tl1b = (const float*)d_in[16];
  const float* tl2g = (const float*)d_in[17];
  const float* tl2b = (const float*)d_in[18];
  const float* tfW1 = (const float*)d_in[19];
  const float* tfb1 = (const float*)d_in[20];
  const float* tfW2 = (const float*)d_in[21];
  const float* tfb2 = (const float*)d_in[22];
  const float* sWq  = (const float*)d_in[23];
  const float* sWk  = (const float*)d_in[24];
  const float* sWv  = (const float*)d_in[25];
  const float* sl1g = (const float*)d_in[26];
  const float* sl1b = (const float*)d_in[27];
  const float* sl2g = (const float*)d_in[28];
  const float* sl2b = (const float*)d_in[29];
  const float* sfW1 = (const float*)d_in[30];
  const float* sfb1 = (const float*)d_in[31];
  const float* sfW2 = (const float*)d_in[32];
  const float* sfb2 = (const float*)d_in[33];
  const float* oW   = (const float*)d_in[35];
  const float* ob   = (const float*)d_in[36];
  float* out = (float*)d_out;

  char* w = (char*)d_ws;
  __bf16* hn    = (__bf16*)(w + OFF_HN);
  float*  xlf   = (float*)(w + OFF_XLF);
  __bf16* xlbf  = (__bf16*)(w + OFF_XLBF);
  float*  xln   = (float*)(w + OFF_XLN);
  float*  cv    = (float*)(w + OFF_CV);
  int*    ci    = (int*)(w + OFF_CI);
  float*  tw    = (float*)(w + OFF_TW);
  int*    tix   = (int*)(w + OFF_TI);
  float*  aug   = (float*)(w + OFF_AUG);
  float*  zn2   = (float*)(w + OFF_ZN2);
  float*  qb    = (float*)(w + OFF_Q);
  float*  kb    = (float*)(w + OFF_K);
  float*  vb    = (float*)(w + OFF_V);
  float*  sc    = (float*)(w + OFF_SC);
  float*  att   = (float*)(w + OFF_ATT);

  hist_prep<<<(NPADH + 7) / 8, 256, 0, stream>>>(hist, hn);
  xl_prep<<<NB / 8, 256, 0, stream>>>(x, xlf, xlbf, xln);
  sims_topk<<<dim3(NB / 16, NCHUNKS), 256, 0, stream>>>(hn, xlbf, cv, ci);
  topk_merge<<<NB, 32, 0, stream>>>(cv, ci, xln, tw, tix);
  build_aug<<<NB, 256, 0, stream>>>(hist, xlf, tw, tix, aug);
  mlp_chain<<<NB / 8, 256, 0, stream>>>(aug, x, fW1, fb1, fW2, fb2, flg, flb, gW, gb,
                                        inW, inb, tWv, tl1g, tl1b, tl2g, tl2b,
                                        tfW1, tfb1, tfW2, tfb2, sl1g, sl1b, zn2);
  qkv_wmma<<<NB / 16, 256, 0, stream>>>(zn2, sWq, sWk, sWv, qb, kb, vb);
  scores_wmma<<<dim3(64, 2), 256, 0, stream>>>(qb, kb, sc);
  softmax1024<<<2048, 256, 0, stream>>>(sc);
  av_wmma<<<dim3(64, 2), 256, 0, stream>>>(sc, vb, att);
  final_kernel<<<NB / 8, 256, 0, stream>>>(zn2, att, sl2g, sl2b, sfW1, sfb1, sfW2, sfb2, oW, ob, out);
}
